// NumFeaturesEncoder_87479893884945
// MI455X (gfx1250) — matrix operand order repaired and checked by reading, not compiled, not run on hardware
//
#include <hip/hip_runtime.h>
#include <hip/hip_bf16.h>

typedef __bf16 bf16;
typedef __attribute__((ext_vector_type(8)))  __bf16 v8bf;
typedef __attribute__((ext_vector_type(16))) __bf16 v16bf;
typedef __attribute__((ext_vector_type(8)))  float  v8f;
typedef __attribute__((ext_vector_type(4)))  int    v4i;

// ---------------------------------------------------------------- constants
#define KB_B   65536
#define KB_D0  256
#define KB_D1  512
#define KB_D2  256
#define KB_K0  (9 * KB_D0)   // 2304
#define KB_K1  (9 * KB_D1)   // 4608
#define KB_CH  8192          // batch chunk: keeps A-slices L2-resident (192MB L2)

#define BM 128
#define BN 128
#define BK 32
#define LDSP (BK + 8)        // +16B pad per row against LDS bank conflicts

// CDNA5 async global->LDS copy (no VGPR round-trip, tracked by ASYNCcnt)
#if defined(__HIP_DEVICE_COMPILE__) && __has_builtin(__builtin_amdgcn_global_load_async_to_lds_b128)
#define USE_ASYNC_LDS 1
#else
#define USE_ASYNC_LDS 0
#endif

#if USE_ASYNC_LDS
// builtin signature: (v4i addrspace(1)* src, v4i addrspace(3)* dst, imm off, imm cpol)
#define ASYNC_CP_B128(gp, lp)                                                  \
    __builtin_amdgcn_global_load_async_to_lds_b128(                            \
        (__attribute__((address_space(1))) v4i*)(v4i*)(void*)(gp),             \
        (__attribute__((address_space(3))) v4i*)(v4i*)(void*)(lp), 0, 0)

__device__ __forceinline__ void wait_async0() {
#if __has_builtin(__builtin_amdgcn_s_wait_asynccnt)
    __builtin_amdgcn_s_wait_asynccnt(0);
#else
    asm volatile("s_wait_asynccnt 0" ::: "memory");
#endif
}
#endif

// ---------------------------------------------------------------- activations
__device__ __forceinline__ float gelu_tanh(float x) {
    const float k0 = 0.7978845608028654f;   // sqrt(2/pi)
    float x3 = x * x * x;
    return 0.5f * x * (1.0f + tanhf(k0 * (x + 0.044715f * x3)));
}

// Cubic B-spline bases on efficient-kan grid: knots t_j = -2.2 + 0.4*j, j=0..11
__device__ __forceinline__ void bspline8(float x, float* __restrict__ out) {
    float b[11];
#pragma unroll
    for (int j = 0; j < 11; ++j) {
        float tj = -2.2f + 0.4f * (float)j;
        b[j] = (x >= tj && x < tj + 0.4f) ? 1.0f : 0.0f;
    }
#pragma unroll
    for (int k = 1; k <= 3; ++k) {
        float inv = 1.0f / (0.4f * (float)k);
#pragma unroll
        for (int j = 0; j < 11 - k; ++j) {
            float tj   = -2.2f + 0.4f * (float)j;
            float tjk1 = tj + 0.4f * (float)(k + 1);
            b[j] = (x - tj) * inv * b[j] + (tjk1 - x) * inv * b[j + 1];
        }
    }
#pragma unroll
    for (int j = 0; j < 8; ++j) out[j] = b[j];
}

// Build augmented activation: Aout[row][j*IN + i], j=0 gelu, j=1..8 bases.
__global__ __launch_bounds__(256) void kan_act(const float* __restrict__ X,
                                               bf16* __restrict__ Aout,
                                               int rows, int IN, int Ktot) {
    int idx = blockIdx.x * 256 + threadIdx.x;
    if (idx >= rows * IN) return;
    int row = idx / IN;
    int i   = idx - row * IN;
    float v = X[(size_t)row * IN + i];
    float bb[8];
    bspline8(v, bb);
    bf16* dst = Aout + (size_t)row * Ktot;
    dst[i] = (bf16)gelu_tanh(v);          // 9 coalesced column-block streams
#pragma unroll
    for (int g = 0; g < 8; ++g) dst[(size_t)(g + 1) * IN + i] = (bf16)bb[g];
}

// Augmented weights, N-major (OUT x Ktot) bf16: column of B == contiguous row.
__global__ __launch_bounds__(256) void kan_pack_w(const float* __restrict__ baseW,
                                                  const float* __restrict__ splineW,
                                                  const float* __restrict__ splineS,
                                                  bf16* __restrict__ Wp,
                                                  int OUT, int IN) {
    int Ktot = 9 * IN;
    int idx = blockIdx.x * 256 + threadIdx.x;
    if (idx >= OUT * Ktot) return;
    int o = idx / Ktot;
    int k = idx - o * Ktot;
    int j = k / IN;
    int i = k - j * IN;
    float w;
    if (j == 0) w = baseW[(size_t)o * IN + i];
    else        w = splineW[((size_t)o * IN + i) * 8 + (j - 1)] * splineS[(size_t)o * IN + i];
    Wp[idx] = (bf16)w;
}

// ---------------------------------------------------------------- bf16 WMMA GEMM
// C[M,N] = A[M,K] * Bw[N,K]^T   (A row-major bf16, Bw N-major bf16, C f32)
__global__ __launch_bounds__(256) void kan_gemm_bf16(const bf16* __restrict__ A,
                                                     const bf16* __restrict__ Bw,
                                                     float* __restrict__ C,
                                                     int M, int N, int K) {
    __shared__ alignas(16) bf16 As[2][BM][LDSP];
    __shared__ alignas(16) bf16 Bs[2][BN][LDSP];

    const int tid  = threadIdx.x;
    const int lane = tid & 31;
    const int wave = tid >> 5;       // 8 waves (wave32)
    const int wm   = wave & 1;       // 2 waves along M  -> 64 rows each
    const int wn   = wave >> 1;      // 4 waves along N  -> 32 cols each
    const int fr   = lane & 15;
    const int hw   = lane >> 4;      // lane half
    const int fka  = hw << 3;        // A frag: K 0-7/16-23 vs 8-15/24-31
    const int fkb  = hw << 3;        // B frag: K 0-7/16-23 vs 8-15/24-31
    const int bm0  = blockIdx.y * BM;
    const int bn0  = blockIdx.x * BN;

    const int r0 = tid >> 2;         // staging: 64 rows per pass
    const int kc = (tid & 3) << 3;   // 8 bf16 (16B) per load

    // per-thread staging source rows (fixed for whole kernel)
    const bf16* ga0 = A  + (size_t)(bm0 + r0)      * K + kc;
    const bf16* ga1 = A  + (size_t)(bm0 + r0 + 64) * K + kc;
    const bf16* gb0 = Bw + (size_t)(bn0 + r0)      * K + kc;
    const bf16* gb1 = Bw + (size_t)(bn0 + r0 + 64) * K + kc;

    v8f acc[4][2];
#pragma unroll
    for (int i = 0; i < 4; ++i)
#pragma unroll
        for (int j = 0; j < 2; ++j)
            acc[i][j] = v8f{0.f, 0.f, 0.f, 0.f, 0.f, 0.f, 0.f, 0.f};

    const int nk = K / BK;

    // ---- stage k-tile 0
#if USE_ASYNC_LDS
    ASYNC_CP_B128(ga0, &As[0][r0][kc]);
    ASYNC_CP_B128(ga1, &As[0][r0 + 64][kc]);
    ASYNC_CP_B128(gb0, &Bs[0][r0][kc]);
    ASYNC_CP_B128(gb1, &Bs[0][r0 + 64][kc]);
    wait_async0();
#else
    {
        uint4 ta0 = *(const uint4*)ga0;
        uint4 ta1 = *(const uint4*)ga1;
        uint4 tb0 = *(const uint4*)gb0;
        uint4 tb1 = *(const uint4*)gb1;
        *(uint4*)&As[0][r0][kc]      = ta0;
        *(uint4*)&As[0][r0 + 64][kc] = ta1;
        *(uint4*)&Bs[0][r0][kc]      = tb0;
        *(uint4*)&Bs[0][r0 + 64][kc] = tb1;
    }
#endif
    __syncthreads();

    for (int kt = 0; kt < nk; ++kt) {
        const int cur = kt & 1;
        if (kt + 1 < nk) {
            const int nxt  = cur ^ 1;
            const size_t koff = (size_t)(kt + 1) * BK;
#if USE_ASYNC_LDS
            ASYNC_CP_B128(ga0 + koff, &As[nxt][r0][kc]);
            ASYNC_CP_B128(ga1 + koff, &As[nxt][r0 + 64][kc]);
            ASYNC_CP_B128(gb0 + koff, &Bs[nxt][r0][kc]);
            ASYNC_CP_B128(gb1 + koff, &Bs[nxt][r0 + 64][kc]);
#else
            uint4 ta0 = *(const uint4*)(ga0 + koff);
            uint4 ta1 = *(const uint4*)(ga1 + koff);
            uint4 tb0 = *(const uint4*)(gb0 + koff);
            uint4 tb1 = *(const uint4*)(gb1 + koff);
            *(uint4*)&As[nxt][r0][kc]      = ta0;
            *(uint4*)&As[nxt][r0 + 64][kc] = ta1;
            *(uint4*)&Bs[nxt][r0][kc]      = tb0;
            *(uint4*)&Bs[nxt][r0 + 64][kc] = tb1;
#endif
        }

        v16bf af[4], bfr[2];
#pragma unroll
        for (int i = 0; i < 4; ++i) {
            int row = wm * 64 + i * 16 + fr;
            v8bf lo = *(const v8bf*)&As[cur][row][fka];
            v8bf hi = *(const v8bf*)&As[cur][row][fka + 16];
            af[i] = __builtin_shufflevector(lo, hi, 0, 1, 2, 3, 4, 5, 6, 7,
                                            8, 9, 10, 11, 12, 13, 14, 15);
        }
#pragma unroll
        for (int j = 0; j < 2; ++j) {
            int col = wn * 32 + j * 16 + fr;
            v8bf lo = *(const v8bf*)&Bs[cur][col][fkb];
            v8bf hi = *(const v8bf*)&Bs[cur][col][fkb + 16];
            bfr[j] = __builtin_shufflevector(lo, hi, 0, 1, 2, 3, 4, 5, 6, 7,
                                             8, 9, 10, 11, 12, 13, 14, 15);
        }
#pragma unroll
        for (int i = 0; i < 4; ++i)
#pragma unroll
            for (int j = 0; j < 2; ++j)
                acc[i][j] = __builtin_amdgcn_wmma_f32_16x16x32_bf16(
                    false, af[i], false, bfr[j], (short)0, acc[i][j], false, false);

#if USE_ASYNC_LDS
        wait_async0();        // next buffer fully in LDS before the barrier
#endif
        __syncthreads();
    }

    // C/D layout: VGPR r -> M = r + 8*(lane>=16), lane&15 -> N
#pragma unroll
    for (int i = 0; i < 4; ++i) {
        int mbase = bm0 + wm * 64 + i * 16 + hw * 8;
#pragma unroll
        for (int j = 0; j < 2; ++j) {
            int n = bn0 + wn * 32 + j * 16 + fr;
#pragma unroll
            for (int r = 0; r < 8; ++r)
                C[(size_t)(mbase + r) * N + n] = acc[i][j][r];
        }
    }
}

// ---------------------------------------------------------------- LayerNorm (D=256)
// one wave32 per row, 8 f32 per lane, in-place on d_out
__global__ __launch_bounds__(256) void kan_ln(float* __restrict__ out,
                                              const float* __restrict__ g,
                                              const float* __restrict__ b) {
    const int lane = threadIdx.x & 31;
    const int wave = threadIdx.x >> 5;
    size_t row = (size_t)blockIdx.x * 8 + wave;
    float* p = out + row * 256;
    float4 v0 = *(float4*)&p[lane * 8];
    float4 v1 = *(float4*)&p[lane * 8 + 4];
    float s  = v0.x + v0.y + v0.z + v0.w + v1.x + v1.y + v1.z + v1.w;
    float ss = v0.x * v0.x + v0.y * v0.y + v0.z * v0.z + v0.w * v0.w
             + v1.x * v1.x + v1.y * v1.y + v1.z * v1.z + v1.w * v1.w;
#pragma unroll
    for (int m = 16; m > 0; m >>= 1) {
        s  += __shfl_xor(s,  m, 32);
        ss += __shfl_xor(ss, m, 32);
    }
    float mean = s * (1.0f / 256.0f);
    float var  = ss * (1.0f / 256.0f) - mean * mean;
    float rs   = rsqrtf(var + 1e-5f);
    float4 g0 = *(const float4*)&g[lane * 8];
    float4 g1 = *(const float4*)&g[lane * 8 + 4];
    float4 b0 = *(const float4*)&b[lane * 8];
    float4 b1 = *(const float4*)&b[lane * 8 + 4];
    float4 o0, o1;
    o0.x = (v0.x - mean) * rs * g0.x + b0.x;
    o0.y = (v0.y - mean) * rs * g0.y + b0.y;
    o0.z = (v0.z - mean) * rs * g0.z + b0.z;
    o0.w = (v0.w - mean) * rs * g0.w + b0.w;
    o1.x = (v1.x - mean) * rs * g1.x + b1.x;
    o1.y = (v1.y - mean) * rs * g1.y + b1.y;
    o1.z = (v1.z - mean) * rs * g1.z + b1.z;
    o1.w = (v1.w - mean) * rs * g1.w + b1.w;
    *(float4*)&p[lane * 8]     = o0;
    *(float4*)&p[lane * 8 + 4] = o1;
}

// ---------------------------------------------------------------- launcher
extern "C" void kernel_launch(void* const* d_in, const int* in_sizes, int n_in,
                              void* d_out, int out_size, void* d_ws, size_t ws_size,
                              hipStream_t stream) {
    (void)in_sizes; (void)n_in; (void)out_size; (void)ws_size;
    const float* x         = (const float*)d_in[0];
    const float* base_w0   = (const float*)d_in[1];
    const float* spline_w0 = (const float*)d_in[2];
    const float* spline_s0 = (const float*)d_in[3];
    const float* base_w1   = (const float*)d_in[4];
    const float* spline_w1 = (const float*)d_in[5];
    const float* spline_s1 = (const float*)d_in[6];
    const float* ln_g      = (const float*)d_in[7];
    const float* ln_b      = (const float*)d_in[8];
    float* out = (float*)d_out;

    // workspace layout (~135 MB)
    bf16* W0p = (bf16*)d_ws;                        // D1 x K0
    bf16* W1p = W0p + (size_t)KB_D1 * KB_K0;        // D2 x K1
    bf16* A0  = W1p + (size_t)KB_D2 * KB_K1;        // CH x K0
    bf16* A1  = A0  + (size_t)KB_CH * KB_K0;        // CH x K1
    float* H  = (float*)(A1 + (size_t)KB_CH * KB_K1); // CH x D1

    kan_pack_w<<<(KB_D1 * KB_K0 + 255) / 256, 256, 0, stream>>>(
        base_w0, spline_w0, spline_s0, W0p, KB_D1, KB_D0);
    kan_pack_w<<<(KB_D2 * KB_K1 + 255) / 256, 256, 0, stream>>>(
        base_w1, spline_w1, spline_s1, W1p, KB_D2, KB_D1);

    for (int c = 0; c < KB_B / KB_CH; ++c) {
        const size_t r0 = (size_t)c * KB_CH;
        kan_act<<<(KB_CH * KB_D0 + 255) / 256, 256, 0, stream>>>(
            x + r0 * KB_D0, A0, KB_CH, KB_D0, KB_K0);
        kan_gemm_bf16<<<dim3(KB_D1 / BN, KB_CH / BM), 256, 0, stream>>>(
            A0, W0p, H, KB_CH, KB_D1, KB_K0);
        kan_act<<<(KB_CH * KB_D1 + 255) / 256, 256, 0, stream>>>(
            H, A1, KB_CH, KB_D1, KB_K1);
        kan_gemm_bf16<<<dim3(KB_D2 / BN, KB_CH / BM), 256, 0, stream>>>(
            A1, W1p, out + r0 * KB_D2, KB_CH, KB_D2, KB_K1);
        kan_ln<<<KB_CH / 8, 256, 0, stream>>>(out + r0 * KB_D2, ln_g, ln_b);
    }
}
